// SiglipVideoTextMatcher_22179211117426
// MI455X (gfx1250) — matrix operand order repaired and checked by reading, not compiled, not run on hardware
//
#include <hip/hip_runtime.h>
#include <hip/hip_bf16.h>
#include <math.h>

#define H_   768
#define NH_  12
#define S_   2048
#define B_   2
#define HD_  64
#define MLP_ 3072
#define ROWS (B_ * S_)          // 4096
#define NEG_INF (-__builtin_inff())

typedef __attribute__((ext_vector_type(16))) __bf16 v16bf;
typedef __attribute__((ext_vector_type(8)))  float  v8f;

union Frag32 { uint4 q[2]; v16bf v; };

__device__ inline __bf16 f2bf(float f) {
    unsigned int u = __builtin_bit_cast(unsigned int, f);
    unsigned int r = (u + 0x7FFFu + ((u >> 16) & 1u)) >> 16;
    unsigned short s = (unsigned short)r;
    return __builtin_bit_cast(__bf16, s);
}
__device__ inline v16bf ldfrag(const __bf16* p0, const __bf16* p1) {
    Frag32 f;
    f.q[0] = *reinterpret_cast<const uint4*>(p0);
    f.q[1] = *reinterpret_cast<const uint4*>(p1);
    return f.v;
}
__device__ inline float redmax16(float v) {
    v = fmaxf(v, __shfl_xor(v, 1, 32));
    v = fmaxf(v, __shfl_xor(v, 2, 32));
    v = fmaxf(v, __shfl_xor(v, 4, 32));
    v = fmaxf(v, __shfl_xor(v, 8, 32));
    return v;
}
__device__ inline float redsum16(float v) {
    v += __shfl_xor(v, 1, 32);
    v += __shfl_xor(v, 2, 32);
    v += __shfl_xor(v, 4, 32);
    v += __shfl_xor(v, 8, 32);
    return v;
}

// ---------------- fp32 [K,N] -> bf16 [N,K] convert+transpose (one-time) ----
__global__ __launch_bounds__(256)
void svtm_cvt_tr_kernel(const float* __restrict__ src, __bf16* __restrict__ dst,
                        int K, int N) {
    int i = blockIdx.x * 256 + threadIdx.x;
    if (i >= K * N) return;
    int k = i / N, n = i % N;
    dst[(size_t)n * K + k] = f2bf(src[i]);
}

// ---------------- LayerNorm (fp32 in, bf16 out) ----------------
__global__ __launch_bounds__(256)
void svtm_ln_kernel(const float* __restrict__ x, const float* __restrict__ g,
                    const float* __restrict__ bta, __bf16* __restrict__ out) {
    __shared__ float red[256];
    const int row = blockIdx.x;
    const int t = threadIdx.x;
    const float* xr = x + (size_t)row * H_;
    float v0 = xr[t], v1 = xr[t + 256], v2 = xr[t + 512];
    red[t] = v0 + v1 + v2;
    __syncthreads();
    for (int o = 128; o > 0; o >>= 1) { if (t < o) red[t] += red[t + o]; __syncthreads(); }
    float mu = red[0] * (1.0f / H_);
    __syncthreads();
    float d0 = v0 - mu, d1 = v1 - mu, d2 = v2 - mu;
    red[t] = d0 * d0 + d1 * d1 + d2 * d2;
    __syncthreads();
    for (int o = 128; o > 0; o >>= 1) { if (t < o) red[t] += red[t + o]; __syncthreads(); }
    float rs = rsqrtf(red[0] * (1.0f / H_) + 1e-5f);
    __bf16* orow = out + (size_t)row * H_;
    orow[t]       = f2bf(d0 * rs * g[t]       + bta[t]);
    orow[t + 256] = f2bf(d1 * rs * g[t + 256] + bta[t + 256]);
    orow[t + 512] = f2bf(d2 * rs * g[t + 512] + bta[t + 512]);
}

// ---------------- LDS-free bf16 WMMA GEMM: C = act(A@W + bias [+ resid]) ---
// A: [M,K] bf16 row-major, Wt: [N,K] bf16 (pre-transposed). Block = 128 thr
// (4 waves), 64x64 C tile; wave w owns rows w*16..w*16+15. All fragments are
// contiguous 2x16B global loads in the same k = hb*16 + j order for A and B, so the
// inner loop is pure global_load_b128 + v_wmma with no LDS/barriers.
template <int KC>
__global__ __launch_bounds__(128)
void svtm_gemm_kernel(const __bf16* __restrict__ A, const __bf16* __restrict__ Wt,
                      const float* __restrict__ bias, const float* __restrict__ resid,
                      float* __restrict__ outF, __bf16* __restrict__ outB,
                      int M, int N, int act) {
    const int lane = threadIdx.x & 31, wid = threadIdx.x >> 5;
    const int l15 = lane & 15, hb = lane >> 4;
    const int n0 = blockIdx.x * 64, m0 = blockIdx.y * 64;

    v8f acc[4];
    v8f zero = {0.f, 0.f, 0.f, 0.f, 0.f, 0.f, 0.f, 0.f};
    acc[0] = zero; acc[1] = zero; acc[2] = zero; acc[3] = zero;

    // A frag chunks: element j -> k = hb*16 + j
    const __bf16* ap = A + (size_t)(m0 + wid * 16 + l15) * KC + hb * 16;
    // B frag chunks: element j -> k = hb*16 + j
    const __bf16* bp0 = Wt + (size_t)(n0 + l15) * KC + hb * 16;

#pragma unroll 2
    for (int k0 = 0; k0 < KC; k0 += 32) {
        v16bf af = ldfrag(ap + k0, ap + k0 + 8);
        v16bf b0 = ldfrag(bp0 + k0,               bp0 + k0 + 8);
        v16bf b1 = ldfrag(bp0 + 16 * KC + k0,     bp0 + 16 * KC + k0 + 8);
        v16bf b2 = ldfrag(bp0 + 32 * KC + k0,     bp0 + 32 * KC + k0 + 8);
        v16bf b3 = ldfrag(bp0 + 48 * KC + k0,     bp0 + 48 * KC + k0 + 8);
        acc[0] = __builtin_amdgcn_wmma_f32_16x16x32_bf16(false, af, false, b0, (short)0, acc[0], false, false);
        acc[1] = __builtin_amdgcn_wmma_f32_16x16x32_bf16(false, af, false, b1, (short)0, acc[1], false, false);
        acc[2] = __builtin_amdgcn_wmma_f32_16x16x32_bf16(false, af, false, b2, (short)0, acc[2], false, false);
        acc[3] = __builtin_amdgcn_wmma_f32_16x16x32_bf16(false, af, false, b3, (short)0, acc[3], false, false);
    }
    // epilogue: C layout -> row = r + hb*8, col = lane&15 within each 16x16 tile
#pragma unroll
    for (int n = 0; n < 4; ++n) {
        int col = n0 + n * 16 + l15;
        float bv = bias ? bias[col] : 0.f;
#pragma unroll
        for (int r = 0; r < 8; ++r) {
            int row = m0 + wid * 16 + r + hb * 8;
            float v = acc[n][r] + bv;
            if (resid) v += resid[(size_t)row * N + col];
            if (act) v = 0.5f * v * (1.0f + erff(v * 0.70710678118654752f));
            if (outF) outF[(size_t)row * N + col] = v;
            if (outB) outB[(size_t)row * N + col] = f2bf(v);
        }
    }
}

// ---------------- RoPE + head-major repack ----------------
// q,k -> [B,NH,S,HD] bf16 (with rotation); v -> [B,NH,HD,S] bf16 (transposed)
__global__ __launch_bounds__(256)
void svtm_rope_pack_kernel(const float* __restrict__ q_lin, const float* __restrict__ k_lin,
                           const float* __restrict__ v_lin,
                           __bf16* __restrict__ qh, __bf16* __restrict__ kh,
                           __bf16* __restrict__ vt) {
    int i = blockIdx.x * 256 + threadIdx.x;
    const int total = ROWS * H_;
    if (i >= total) return;
    int col = i % H_;
    int row = i / H_;
    int sp = row % S_;
    int b  = row / S_;
    int h  = col / HD_;
    int d  = col % HD_;
    float inv = powf(10000.0f, -(float)(2 * (d & 31)) / (float)HD_);
    float ang = (float)sp * inv;
    float c = cosf(ang), sn = sinf(ang);
    float qv = q_lin[i], kv = k_lin[i];
    float qo, ko;
    if (d < 32) { qo = qv * c - q_lin[i + 32] * sn; ko = kv * c - k_lin[i + 32] * sn; }
    else        { qo = qv * c + q_lin[i - 32] * sn; ko = kv * c + k_lin[i - 32] * sn; }
    size_t bh = (size_t)(b * NH_ + h);
    qh[(bh * S_ + sp) * HD_ + d] = f2bf(qo);
    kh[(bh * S_ + sp) * HD_ + d] = f2bf(ko);
    vt[(bh * HD_ + d) * S_ + sp] = f2bf(v_lin[i]);
}

// ---------------- flash attention (bf16 WMMA, fp32 online softmax) ------
// grid = (S/64, B*NH), block = 128 (4 waves); wave -> 16 q rows.
__global__ __launch_bounds__(128)
void svtm_attn_kernel(const __bf16* __restrict__ qh, const __bf16* __restrict__ kh,
                      const __bf16* __restrict__ vt, const int* __restrict__ amask,
                      __bf16* __restrict__ ctx) {
    __shared__ __align__(16) __bf16 Pbuf[4][16 * 40];
    const int lane = threadIdx.x & 31, wid = threadIdx.x >> 5;
    const int l15 = lane & 15, hb = lane >> 4;
    const int bh = blockIdx.y;
    const int b = bh / NH_, h = bh % NH_;
    const int q0 = blockIdx.x * 64 + wid * 16;
    const size_t base = (size_t)bh * S_ * HD_;
    const int ab = hb * 16;

    // q A-fragments for K=0..31 and 32..63
    const __bf16* qp = qh + base + (size_t)(q0 + l15) * HD_;
    v16bf qf0 = ldfrag(qp + ab,      qp + ab + 8);
    v16bf qf1 = ldfrag(qp + ab + 32, qp + ab + 40);

    float mrow[8], lrow[8];
    v8f o[4];
    v8f zero = {0.f, 0.f, 0.f, 0.f, 0.f, 0.f, 0.f, 0.f};
#pragma unroll
    for (int r = 0; r < 8; ++r) { mrow[r] = NEG_INF; lrow[r] = 0.f; }
    o[0] = zero; o[1] = zero; o[2] = zero; o[3] = zero;

    const int* mp = amask + b * S_;
    __bf16* P = &Pbuf[wid][0];

    for (int kt = 0; kt < S_; kt += 32) {
        v8f s0 = zero, s1 = zero;
        const __bf16* kp0 = kh + base + (size_t)(kt + l15) * HD_ + hb * 16;
        const __bf16* kp1 = kh + base + (size_t)(kt + 16 + l15) * HD_ + hb * 16;
        v16bf b00 = ldfrag(kp0,      kp0 + 8);   // feat 0..31
        v16bf b10 = ldfrag(kp1,      kp1 + 8);
        v16bf b01 = ldfrag(kp0 + 32, kp0 + 40);  // feat 32..63
        v16bf b11 = ldfrag(kp1 + 32, kp1 + 40);
        s0 = __builtin_amdgcn_wmma_f32_16x16x32_bf16(false, qf0, false, b00, (short)0, s0, false, false);
        s1 = __builtin_amdgcn_wmma_f32_16x16x32_bf16(false, qf0, false, b10, (short)0, s1, false, false);
        s0 = __builtin_amdgcn_wmma_f32_16x16x32_bf16(false, qf1, false, b01, (short)0, s0, false, false);
        s1 = __builtin_amdgcn_wmma_f32_16x16x32_bf16(false, qf1, false, b11, (short)0, s1, false, false);

        bool m0v = mp[kt + l15] != 0;
        bool m1v = mp[kt + 16 + l15] != 0;
        float al[8];
#pragma unroll
        for (int r = 0; r < 8; ++r) {
            float a = m0v ? s0[r] * 0.125f : NEG_INF;
            float c = m1v ? s1[r] * 0.125f : NEG_INF;
            float tm = redmax16(fmaxf(a, c));
            float newM = fmaxf(mrow[r], tm);
            float alpha = (mrow[r] == NEG_INF) ? 0.f : __expf(mrow[r] - newM);
            float p0 = (a == NEG_INF) ? 0.f : __expf(a - newM);
            float p1 = (c == NEG_INF) ? 0.f : __expf(c - newM);
            s0[r] = p0; s1[r] = p1;
            lrow[r] = lrow[r] * alpha + redsum16(p0 + p1);
            mrow[r] = newM;
            al[r] = alpha;
        }
#pragma unroll
        for (int n = 0; n < 4; ++n)
#pragma unroll
            for (int r = 0; r < 8; ++r) o[n][r] *= al[r];

        // P (16q x 32k) C-layout -> LDS -> A-fragment layout
#pragma unroll
        for (int r = 0; r < 8; ++r) {
            P[(r + hb * 8) * 40 + l15]      = f2bf(s0[r]);
            P[(r + hb * 8) * 40 + 16 + l15] = f2bf(s1[r]);
        }
        asm volatile("s_wait_dscnt 0" ::: "memory");
        v16bf pf = ldfrag(&P[l15 * 40 + ab], &P[l15 * 40 + ab + 8]);

        const __bf16* vp = vt + (size_t)bh * HD_ * S_ + kt + hb * 16;
#pragma unroll
        for (int n = 0; n < 4; ++n) {
            const __bf16* vpp = vp + (size_t)(n * 16 + l15) * S_;
            v16bf vf = ldfrag(vpp, vpp + 8);
            o[n] = __builtin_amdgcn_wmma_f32_16x16x32_bf16(
                false, pf, false, vf, (short)0, o[n], false, false);
        }
    }
    // normalize + write ctx [B*S, H] row-major bf16
#pragma unroll
    for (int n = 0; n < 4; ++n)
#pragma unroll
        for (int r = 0; r < 8; ++r) {
            int row = q0 + r + hb * 8;
            int col = h * HD_ + n * 16 + l15;
            float v = o[n][r] / lrow[r];
            ctx[((size_t)b * S_ + row) * H_ + col] = f2bf(v);
        }
}

extern "C" void kernel_launch(void* const* d_in, const int* in_sizes, int n_in,
                              void* d_out, int out_size, void* d_ws, size_t ws_size,
                              hipStream_t stream) {
    (void)in_sizes; (void)n_in; (void)out_size; (void)ws_size;
    const float* x     = (const float*)d_in[0];
    const int*   amask = (const int*)d_in[1];
    const float* ln1g  = (const float*)d_in[2];
    const float* ln1b  = (const float*)d_in[3];
    const float* ln2g  = (const float*)d_in[4];
    const float* ln2b  = (const float*)d_in[5];
    const float* wq = (const float*)d_in[6];  const float* bq = (const float*)d_in[7];
    const float* wk = (const float*)d_in[8];  const float* bk = (const float*)d_in[9];
    const float* wv = (const float*)d_in[10]; const float* bv = (const float*)d_in[11];
    const float* wo = (const float*)d_in[12]; const float* bo = (const float*)d_in[13];
    const float* w1 = (const float*)d_in[14]; const float* b1 = (const float*)d_in[15];
    const float* w2 = (const float*)d_in[16]; const float* b2 = (const float*)d_in[17];

    char* wsb = (char*)d_ws;
    size_t off = 0;
    auto alloc = [&](size_t bytes) -> void* {
        void* p = wsb + off;
        off += (bytes + 255) & ~(size_t)255;
        return p;
    };
    __bf16* wq_bf = (__bf16*)alloc((size_t)H_ * H_ * 2);     // [N,K] transposed
    __bf16* wk_bf = (__bf16*)alloc((size_t)H_ * H_ * 2);
    __bf16* wv_bf = (__bf16*)alloc((size_t)H_ * H_ * 2);
    __bf16* wo_bf = (__bf16*)alloc((size_t)H_ * H_ * 2);
    __bf16* w1_bf = (__bf16*)alloc((size_t)H_ * MLP_ * 2);   // [MLP,768]
    __bf16* w2_bf = (__bf16*)alloc((size_t)MLP_ * H_ * 2);   // [768,3072]
    __bf16* a_in  = (__bf16*)alloc((size_t)ROWS * H_ * 2);
    float*  q_lin = (float*)alloc((size_t)ROWS * H_ * 4);   // later aliased as x1
    float*  k_lin = (float*)alloc((size_t)ROWS * H_ * 4);   // later aliased as h (part 1)
    float*  v_lin = (float*)alloc((size_t)ROWS * H_ * 4);   // later aliased as h (part 2)
    __bf16* qh    = (__bf16*)alloc((size_t)ROWS * H_ * 2);
    __bf16* kh    = (__bf16*)alloc((size_t)ROWS * H_ * 2);
    __bf16* vt    = (__bf16*)alloc((size_t)ROWS * H_ * 2);
    __bf16* ctx   = (__bf16*)alloc((size_t)ROWS * H_ * 2);
    __bf16* m_in  = (__bf16*)alloc((size_t)ROWS * H_ * 2);
    float*  x1    = q_lin;                                  // 4096*768*4 bytes, reuse
    __bf16* hbuf  = (__bf16*)k_lin;                         // 4096*3072*2 = k_lin+v_lin

    // weights -> bf16, transposed to [N,K]
    int nw = H_ * H_;
    svtm_cvt_tr_kernel<<<(nw + 255) / 256, 256, 0, stream>>>(wq, wq_bf, H_, H_);
    svtm_cvt_tr_kernel<<<(nw + 255) / 256, 256, 0, stream>>>(wk, wk_bf, H_, H_);
    svtm_cvt_tr_kernel<<<(nw + 255) / 256, 256, 0, stream>>>(wv, wv_bf, H_, H_);
    svtm_cvt_tr_kernel<<<(nw + 255) / 256, 256, 0, stream>>>(wo, wo_bf, H_, H_);
    int nm = H_ * MLP_;
    svtm_cvt_tr_kernel<<<(nm + 255) / 256, 256, 0, stream>>>(w1, w1_bf, H_, MLP_);
    svtm_cvt_tr_kernel<<<(nm + 255) / 256, 256, 0, stream>>>(w2, w2_bf, MLP_, H_);

    // LN1
    svtm_ln_kernel<<<ROWS, 256, 0, stream>>>(x, ln1g, ln1b, a_in);

    // QKV projections
    dim3 gH(H_ / 64, ROWS / 64);
    svtm_gemm_kernel<H_><<<gH, 128, 0, stream>>>(a_in, wq_bf, bq, nullptr, q_lin, nullptr, ROWS, H_, 0);
    svtm_gemm_kernel<H_><<<gH, 128, 0, stream>>>(a_in, wk_bf, bk, nullptr, k_lin, nullptr, ROWS, H_, 0);
    svtm_gemm_kernel<H_><<<gH, 128, 0, stream>>>(a_in, wv_bf, bv, nullptr, v_lin, nullptr, ROWS, H_, 0);

    // RoPE + pack
    svtm_rope_pack_kernel<<<(ROWS * H_) / 256, 256, 0, stream>>>(q_lin, k_lin, v_lin, qh, kh, vt);

    // attention
    dim3 gA(S_ / 64, B_ * NH_);
    svtm_attn_kernel<<<gA, 128, 0, stream>>>(qh, kh, vt, amask, ctx);

    // wo projection + residual -> x1
    svtm_gemm_kernel<H_><<<gH, 128, 0, stream>>>(ctx, wo_bf, bo, x, x1, nullptr, ROWS, H_, 0);

    // LN2
    svtm_ln_kernel<<<ROWS, 256, 0, stream>>>(x1, ln2g, ln2b, m_in);

    // MLP up + GELU (bf16 out)
    dim3 gM(MLP_ / 64, ROWS / 64);
    svtm_gemm_kernel<H_><<<gM, 128, 0, stream>>>(m_in, w1_bf, b1, nullptr, nullptr, hbuf, ROWS, MLP_, 1);

    // MLP down + bias + residual -> d_out (fp32)
    svtm_gemm_kernel<MLP_><<<gH, 128, 0, stream>>>(hbuf, w2_bf, b2, x1, (float*)d_out, nullptr, ROWS, H_, 0);
}
